// VectorQuantizer_80728205295870
// MI455X (gfx1250) — matrix operand order repaired and checked by reading, not compiled, not run on hardware
//
#include <hip/hip_runtime.h>
#include <cstddef>
#include <cstdint>

#define BATCH   32768
#define DIML    256
#define NCODES  8192
#define BETA    0.25f

typedef __attribute__((ext_vector_type(16))) __bf16 v16bf;
typedef __attribute__((ext_vector_type(8)))  float  v8f;
typedef __attribute__((ext_vector_type(4)))  unsigned int u32x4;
typedef __attribute__((ext_vector_type(8)))  int          i32x8;
typedef __attribute__((ext_vector_type(4)))  int          i32x4;

struct alignas(16) B128 { unsigned int w[4]; };
struct alignas(16) F4   { float x, y, z, w; };

union Frag16 { v16bf v; B128 b[2]; };

#if defined(__has_builtin)
#if __has_builtin(__builtin_amdgcn_tensor_load_to_lds) && __has_builtin(__builtin_amdgcn_s_wait_tensorcnt)
#define HAVE_TDM 1
#endif
#endif
#ifndef HAVE_TDM
#define HAVE_TDM 0
#endif

__device__ __forceinline__ unsigned short f32_to_bf16_bits(float f) {
    union { float f; unsigned int u; } x; x.f = f;
    unsigned int r = (x.u + 0x7FFFu + ((x.u >> 16) & 1u)) >> 16;   // RNE
    return (unsigned short)r;
}

// ---------------- Pass 1a: z_e f32 -> bf16 ----------------
__global__ void __launch_bounds__(256) k_convert_ze(const float* __restrict__ z,
                                                    unsigned short* __restrict__ out) {
    const int n4 = (BATCH * DIML) / 4;
    for (int i = blockIdx.x * blockDim.x + threadIdx.x; i < n4;
         i += gridDim.x * blockDim.x) {
        F4 v = ((const F4*)z)[i];
        unsigned int lo = (unsigned int)f32_to_bf16_bits(v.x) |
                          ((unsigned int)f32_to_bf16_bits(v.y) << 16);
        unsigned int hi = (unsigned int)f32_to_bf16_bits(v.z) |
                          ((unsigned int)f32_to_bf16_bits(v.w) << 16);
        unsigned long long p = ((unsigned long long)hi << 32) | lo;
        ((unsigned long long*)out)[i] = p;
    }
}

// ---------------- Pass 1b: codebook -> bf16, e2[k] = ||e_k||^2 ----------------
__global__ void __launch_bounds__(256) k_cb_e2(const float* __restrict__ cb,
                                               unsigned short* __restrict__ cbb,
                                               float* __restrict__ e2) {
    const int k = blockIdx.x;
    const int d = threadIdx.x;
    float v = cb[(size_t)k * DIML + d];
    cbb[(size_t)k * DIML + d] = f32_to_bf16_bits(v);
    float s = v * v;
    #pragma unroll
    for (int o = 16; o > 0; o >>= 1) s += __shfl_xor(s, o, 32);
    __shared__ float red[8];
    if ((d & 31) == 0) red[d >> 5] = s;
    __syncthreads();
    if (d == 0) {
        float t = 0.f;
        #pragma unroll
        for (int i = 0; i < 8; ++i) t += red[i];
        e2[k] = t;
    }
}

// ---------------- Pass 2: fused GEMM + running argmin via bf16 WMMA + TDM staging ----------------
#define KTILE   32                     // codes per LDS tile
#define LDSROW  264                    // 256 bf16 + 8 pad -> conflict-free b128 reads
#define TILEELT (KTILE * LDSROW)       // elements per LDS buffer

#if HAVE_TDM
// Issue one TDM descriptor: 2D tile, KTILE rows x 512 B from the bf16 codebook,
// LDS padding 16 B after every 512 B row (pad_interval=6 -> 128 dwords, pad_amount=3 -> 4 dwords).
__device__ __forceinline__ void tdm_stage(const unsigned short* gsrc, unsigned ldsByteAddr) {
    unsigned long long ga = (unsigned long long)(uintptr_t)gsrc;
    u32x4 g0;
    g0.x = 0x1u;                                            // count=1, user descriptor
    g0.y = ldsByteAddr;                                     // lds_addr [63:32]
    g0.z = (unsigned)(ga & 0xFFFFFFFFu);                    // global_addr lo
    g0.w = (unsigned)((ga >> 32) & 0x01FFFFFFu) | (2u << 30); // global_addr hi | type=2
    i32x8 g1;
    g1[0] = (int)((1u << 16) |                              // data_size = 2 B
                  (1u << 20) |                              // pad_enable
                  (6u << 22) |                              // pad_interval: 128 dwords
                  (3u << 25));                              // pad_amount: 4 dwords
    g1[1] = (int)((unsigned)DIML << 16);                    // tensor_dim0 lo16 (=256)
    g1[2] = (int)((unsigned)NCODES << 16);                  // dim0 hi=0 | tensor_dim1 lo16
    g1[3] = (int)((unsigned)DIML << 16);                    // dim1 hi=0 | tile_dim0 = 256
    g1[4] = KTILE;                                          // tile_dim1 = 32, tile_dim2 = 0
    g1[5] = DIML;                                           // tensor_dim0_stride lo32 = 256
    g1[6] = 0;
    g1[7] = 0;
    i32x4 z4 = {0, 0, 0, 0};                                // groups 2/3 unused (2D)
    i32x8 z8 = {0, 0, 0, 0, 0, 0, 0, 0};                    // 6-arg form: extra group, zero-filled
    __builtin_amdgcn_tensor_load_to_lds(g0, g1, z4, z4, z8, 0);
}
#endif

__global__ void __launch_bounds__(256) k_vq_argmin(
    const unsigned short* __restrict__ zeb,   // [BATCH][DIML] bf16 bits
    const unsigned short* __restrict__ cbb,   // [NCODES][DIML] bf16 bits
    const float* __restrict__ e2,             // [NCODES]
    int* __restrict__ idx_ws,                 // [BATCH]
    float* __restrict__ idx_out)              // [BATCH] as f32
{
    __shared__ __align__(16) unsigned short tile[2 * TILEELT];

    const int tid  = threadIdx.x;
    const int wave = tid >> 5;
    const int lane = tid & 31;
    const int half = lane >> 4;
    const int l16  = lane & 15;
    const int rowBase = blockIdx.x * 128 + wave * 16;

    // A panel: 16 z_e rows x 256 d resident in registers (ISA 16-bit A 16x32 layout).
    Frag16 a[8];
    {
        const unsigned short* zrow = zeb + (size_t)(rowBase + l16) * DIML;
        const int kb = half * 8;
        #pragma unroll
        for (int c = 0; c < 8; ++c) {
            a[c].b[0] = *(const B128*)(zrow + c * 32 + kb);
            a[c].b[1] = *(const B128*)(zrow + c * 32 + 16 + kb);
        }
    }

    float best[8];
    int   bidx[8];
    #pragma unroll
    for (int r = 0; r < 8; ++r) { best[r] = -3.4e38f; bidx[r] = 0; }

#if HAVE_TDM
    const unsigned ldsBase = (unsigned)(uintptr_t)(&tile[0]);  // flat[31:0] == LDS byte offset
    if (wave == 0) tdm_stage(cbb, ldsBase);                    // prologue: tile 0 -> buf 0
#endif

    const int nt = NCODES / KTILE;           // 256 tiles
    for (int t = 0; t < nt; ++t) {
        const int k0 = t * KTILE;
        const unsigned short* buf = tile + (t & 1) * TILEELT;

#if HAVE_TDM
        if (wave == 0) {
            if (t + 1 < nt) {
                tdm_stage(cbb + (size_t)(k0 + KTILE) * DIML,
                          ldsBase + (unsigned)(((t + 1) & 1) * TILEELT * 2));
                __builtin_amdgcn_s_wait_tensorcnt(1);   // tile t done; t+1 in flight
            } else {
                __builtin_amdgcn_s_wait_tensorcnt(0);   // last tile done
            }
        }
        __syncthreads();                                 // publish tile t to all waves
#else
        __syncthreads();
        // Synchronous fallback staging: KTILE rows x 32 chunks of 16 B = 1024 chunks.
        #pragma unroll
        for (int i = 0; i < 4; ++i) {
            int j   = tid + i * 256;
            int r   = j >> 5;
            int c16 = j & 31;
            B128 v = *(const B128*)(cbb + (size_t)(k0 + r) * DIML + c16 * 8);
            *(B128*)((unsigned short*)buf + r * LDSROW + c16 * 8) = v;
        }
        __syncthreads();
#endif

        // Two independent accumulator chains: codes [k0..k0+15] and [k0+16..k0+31].
        const float bias0 = -0.5f * e2[k0 + l16];
        const float bias1 = -0.5f * e2[k0 + 16 + l16];
        v8f c0 = { bias0, bias0, bias0, bias0, bias0, bias0, bias0, bias0 };
        v8f c1 = { bias1, bias1, bias1, bias1, bias1, bias1, bias1, bias1 };

        const unsigned short* b0row = buf + l16 * LDSROW + half * 8;
        const unsigned short* b1row = buf + (16 + l16) * LDSROW + half * 8;
        #pragma unroll
        for (int ch = 0; ch < 8; ++ch) {
            Frag16 b0, b1;
            b0.b[0] = *(const B128*)(b0row + ch * 32);
            b0.b[1] = *(const B128*)(b0row + ch * 32 + 16);
            b1.b[0] = *(const B128*)(b1row + ch * 32);
            b1.b[1] = *(const B128*)(b1row + ch * 32 + 16);
            c0 = __builtin_amdgcn_wmma_f32_16x16x32_bf16(
                     false, a[ch].v, false, b0.v, (short)0, c0, false, false);
            c1 = __builtin_amdgcn_wmma_f32_16x16x32_bf16(
                     false, a[ch].v, false, b1.v, (short)0, c1, false, false);
        }

        const int kc0 = k0 + l16;
        const int kc1 = k0 + 16 + l16;
        #pragma unroll
        for (int r = 0; r < 8; ++r) {
            float s0 = c0[r];
            if (s0 > best[r]) { best[r] = s0; bidx[r] = kc0; }
            float s1 = c1[r];
            if (s1 > best[r]) { best[r] = s1; bidx[r] = kc1; }
        }

        __syncthreads();   // all waves done reading buf before it is re-staged
    }

    // Cross-lane argmax within each 16-lane half (wave32).
    #pragma unroll
    for (int r = 0; r < 8; ++r) {
        float bv = best[r];
        int   bi = bidx[r];
        #pragma unroll
        for (int off = 1; off < 16; off <<= 1) {
            float ov = __shfl_xor(bv, off, 32);
            int   oi = __shfl_xor(bi, off, 32);
            if (ov > bv || (ov == bv && oi < bi)) { bv = ov; bi = oi; }
        }
        if (l16 == 0) {
            int row = rowBase + r + half * 8;
            idx_ws[row]  = bi;
            idx_out[row] = (float)bi;
        }
    }
}

// ---------------- Pass 3: gather z_q + per-row loss sums ----------------
__global__ void __launch_bounds__(256) k_gather(const float* __restrict__ z,
                                                const float* __restrict__ cb,
                                                const int* __restrict__ idx,
                                                float* __restrict__ zq,
                                                float* __restrict__ rowsum) {
    const int row = blockIdx.x;
    const int d   = threadIdx.x;
    const int k   = idx[row];
    float q = cb[(size_t)k * DIML + d];
    zq[(size_t)row * DIML + d] = q;
    float diff = q - z[(size_t)row * DIML + d];
    float s = diff * diff;
    #pragma unroll
    for (int o = 16; o > 0; o >>= 1) s += __shfl_xor(s, o, 32);
    __shared__ float red[8];
    if ((d & 31) == 0) red[d >> 5] = s;
    __syncthreads();
    if (d == 0) {
        float t = 0.f;
        #pragma unroll
        for (int i = 0; i < 8; ++i) t += red[i];
        rowsum[row] = t;
    }
}

// ---------------- Pass 4: deterministic final reduction ----------------
__global__ void __launch_bounds__(256) k_finalize(const float* __restrict__ rowsum,
                                                  float* __restrict__ loss) {
    float s = 0.f;
    for (int i = threadIdx.x; i < BATCH; i += 256) s += rowsum[i];
    #pragma unroll
    for (int o = 16; o > 0; o >>= 1) s += __shfl_xor(s, o, 32);
    __shared__ float red[8];
    if ((threadIdx.x & 31) == 0) red[threadIdx.x >> 5] = s;
    __syncthreads();
    if (threadIdx.x == 0) {
        float t = 0.f;
        #pragma unroll
        for (int i = 0; i < 8; ++i) t += red[i];
        float mean = t * (1.0f / ((float)BATCH * (float)DIML));
        loss[0] = mean * (1.0f + BETA);   // vq_loss
        loss[1] = mean;                   // cb_loss
        loss[2] = mean;                   // commit_loss (== cb_loss numerically)
    }
}

extern "C" void kernel_launch(void* const* d_in, const int* in_sizes, int n_in,
                              void* d_out, int out_size, void* d_ws, size_t ws_size,
                              hipStream_t stream) {
    const float* z_e = (const float*)d_in[0];
    const float* cb  = (const float*)d_in[1];
    float* out = (float*)d_out;

    char* ws = (char*)d_ws;
    unsigned short* ze_bf = (unsigned short*)ws;                                   // 16 MB
    unsigned short* cb_bf = (unsigned short*)(ws + (size_t)BATCH * DIML * 2);      //  4 MB
    float* e2     = (float*)((char*)cb_bf + (size_t)NCODES * DIML * 2);            // 32 KB
    int*   idx    = (int*)  ((char*)e2    + (size_t)NCODES * 4);                   // 128 KB
    float* rowsum = (float*)((char*)idx   + (size_t)BATCH * 4);                    // 128 KB

    float* zq_out   = out;
    float* idx_out  = out + (size_t)BATCH * DIML;
    float* loss_out = idx_out + BATCH;

    k_convert_ze<<<2048, 256, 0, stream>>>(z_e, ze_bf);
    k_cb_e2    <<<NCODES, 256, 0, stream>>>(cb, cb_bf, e2);
    k_vq_argmin<<<BATCH / 128, 256, 0, stream>>>(ze_bf, cb_bf, e2, idx, idx_out);
    k_gather   <<<BATCH, 256, 0, stream>>>(z_e, cb, idx, zq_out, rowsum);
    k_finalize <<<1, 256, 0, stream>>>(rowsum, loss_out);
}
